// GraphAttentionLayer_9423158248227
// MI455X (gfx1250) — compile-verified, matrix operand order checked by reading, not run on hardware
//
#include <hip/hip_runtime.h>
#include <hip/hip_bf16.h>

typedef __attribute__((ext_vector_type(16))) _Float16 v16h;
typedef __attribute__((ext_vector_type(8)))  _Float16 v8h;
typedef __attribute__((ext_vector_type(8)))  float    v8f;
typedef __attribute__((ext_vector_type(4)))  float    f4;
typedef __attribute__((ext_vector_type(4)))  int      i4;

#define IN_F   256
#define OUT_F  64
#define NN     4096
#define ALPHA  0.2f

// ---------------------------------------------------------------------------
// Kernel 1: Wh = h @ W + bias.  One block (64 threads) per (b,n) row.
// Stores WhT as f16 [B, 64, N] (column-major per batch, so WMMA B-tile loads
// are contiguous), and the attention scalars Wh1 = Wh.a1, Wh2 = Wh.a2 (f32).
// ---------------------------------------------------------------------------
__global__ __launch_bounds__(64) void gat_wh_kernel(
    const float* __restrict__ h, const float* __restrict__ W,
    const float* __restrict__ a, const float* __restrict__ bias,
    _Float16* __restrict__ whT, float* __restrict__ wh1,
    float* __restrict__ wh2, int N)
{
    const int row = blockIdx.x;          // b*N + n
    const int b = row / N;
    const int n = row - b * N;
    const int t = threadIdx.x;           // output feature 0..63

    __shared__ float hrow[IN_F];
    __shared__ float red1[OUT_F];
    __shared__ float red2[OUT_F];

    const float* hp = h + (size_t)row * IN_F;
    for (int k = t; k < IN_F; k += 64) hrow[k] = hp[k];
    __syncthreads();

    float acc = bias[t];
#pragma unroll 8
    for (int k = 0; k < IN_F; ++k)
        acc = fmaf(hrow[k], W[k * OUT_F + t], acc);

    whT[((size_t)b * OUT_F + t) * N + n] = (_Float16)acc;

    red1[t] = acc * a[t];
    red2[t] = acc * a[OUT_F + t];
    __syncthreads();
    for (int s = 32; s > 0; s >>= 1) {
        if (t < s) { red1[t] += red1[t + s]; red2[t] += red2[t + s]; }
        __syncthreads();
    }
    if (t == 0) { wh1[row] = red1[0]; wh2[row] = red2[0]; }
}

// ---------------------------------------------------------------------------
// Kernel 2: per-batch max of Wh2 (for the softmax upper bound).
// ---------------------------------------------------------------------------
__global__ __launch_bounds__(256) void gat_max_kernel(
    const float* __restrict__ wh2, float* __restrict__ m2, int N)
{
    const int b = blockIdx.x;
    const int t = threadIdx.x;
    __shared__ float red[256];
    float m = -3.0e38f;
    for (int k = t; k < N; k += 256) m = fmaxf(m, wh2[(size_t)b * N + k]);
    red[t] = m;
    __syncthreads();
    for (int s = 128; s > 0; s >>= 1) {
        if (t < s) red[t] = fmaxf(red[t], red[t + s]);
        __syncthreads();
    }
    if (t == 0) m2[b] = red[0];
}

// ---------------------------------------------------------------------------
// Kernel 3: fused attention.  Block = 4 waves; each block owns a 16-row
// i-tile; wave w covers j in [w*N/4, (w+1)*N/4).  Per 32-wide j-chunk each
// wave builds the f16 A-operand (P tile, exact 16x32 16-bit lane layout),
// loads 4 B-tiles from WhT, and issues 4x v_wmma_f32_16x16x32_f16.
// Fixed per-row max bound => partial accumulators/sums merge by addition.
// ---------------------------------------------------------------------------
__global__ __launch_bounds__(128) void gat_attn_kernel(
    const int* __restrict__ adj, const _Float16* __restrict__ whT,
    const float* __restrict__ wh1, const float* __restrict__ wh2,
    const float* __restrict__ m2, float* __restrict__ out, int N)
{
    const int lane = threadIdx.x & 31;
    const int w    = threadIdx.x >> 5;     // wave id 0..3 (j-split; also nt in merge)
    const int tile = blockIdx.x;           // b * (N/16) + i-tile
    const int b    = tile >> 8;            // N/16 == 256
    const int i0   = (tile & 255) << 4;
    const int r    = lane & 15;
    const int half = lane >> 4;

    __shared__ float ldsC[4 * 4 * 32 * 8]; // [wave][nt][lane][v]  (16 KB)
    __shared__ float ldsS[4 * 32];         // per-lane partial row sums

    const float M2 = m2[b];
    const float w1 = wh1[(size_t)b * N + i0 + r];
    float mrow = w1 + M2;
    mrow = mrow > 0.f ? mrow : ALPHA * mrow;   // leakyrelu(upper bound) >= all e_ij

    const int*      adjrow = adj + ((size_t)b * N + i0 + r) * N;
    const float*    wh2b   = wh2 + (size_t)b * N;
    const _Float16* whTb   = whT + (size_t)b * OUT_F * N;

    const v8f zero8 = {0.f, 0.f, 0.f, 0.f, 0.f, 0.f, 0.f, 0.f};
    v8f c[4];
#pragma unroll
    for (int i = 0; i < 4; ++i) c[i] = zero8;
    // 4 independent row-sum accumulators (breaks the serial add chain so the
    // adds can dual-issue / hide under the WMMAs)
    float ps0 = 0.f, ps1 = 0.f, ps2 = 0.f, ps3 = 0.f;

    const int jspan = N >> 2;              // 1024 per wave
    const int jlo = w * jspan;

    for (int jc = 0; jc < jspan; jc += 32) {
        const int jbase = jlo + jc;
        const int j0 = jbase + half * 8;   // this lane's first K column

        __builtin_prefetch(adjrow + j0 + 128, 0, 0);

        i4 ad[4];
        ad[0] = *(const i4*)(adjrow + j0);
        ad[1] = *(const i4*)(adjrow + j0 + 4);
        ad[2] = *(const i4*)(adjrow + j0 + 16);
        ad[3] = *(const i4*)(adjrow + j0 + 20);
        f4 sv[4];
        sv[0] = *(const f4*)(wh2b + j0);
        sv[1] = *(const f4*)(wh2b + j0 + 4);
        sv[2] = *(const f4*)(wh2b + j0 + 16);
        sv[3] = *(const f4*)(wh2b + j0 + 20);

        // A-operand (P tile): element e<8 -> K = 8*half+e ; e>=8 -> K = 16+8*half+(e-8)
        v16h af;
        float pp[16];
#pragma unroll
        for (int e = 0; e < 16; ++e) {
            float ev = w1 + sv[e >> 2][e & 3];
            ev = ev > 0.f ? ev : ALPHA * ev;          // LeakyReLU
            float px = __expf(ev - mrow);             // <= 1
            px = (ad[e >> 2][e & 3] > 0) ? px : 0.f;  // adjacency mask
            pp[e] = px;
            af[e] = (_Float16)px;
        }
        // 4-way tree accumulation of the row sum
        ps0 += (pp[0] + pp[4]) + (pp[8]  + pp[12]);
        ps1 += (pp[1] + pp[5]) + (pp[9]  + pp[13]);
        ps2 += (pp[2] + pp[6]) + (pp[10] + pp[14]);
        ps3 += (pp[3] + pp[7]) + (pp[11] + pp[15]);

        // 4 B-tiles (WhT rows nt*16+r), same K-lane layout as A
#pragma unroll
        for (int nt = 0; nt < 4; ++nt) {
            const _Float16* bp = whTb + (size_t)(nt * 16 + r) * N + j0;
            v8h blo = *(const v8h*)bp;
            v8h bhi = *(const v8h*)(bp + 16);
            v16h bf = __builtin_shufflevector(blo, bhi,
                0, 1, 2, 3, 4, 5, 6, 7, 8, 9, 10, 11, 12, 13, 14, 15);
            c[nt] = __builtin_amdgcn_wmma_f32_16x16x32_f16(
                false, af, false, bf, (short)0, c[nt], false, false);
        }
    }
    const float psum = (ps0 + ps1) + (ps2 + ps3);

    // ---- merge the 4 j-split partials through LDS (pure addition) ----
#pragma unroll
    for (int nt = 0; nt < 4; ++nt)
#pragma unroll
        for (int v = 0; v < 8; ++v)
            ldsC[(((w * 4 + nt) * 32) + lane) * 8 + v] = c[nt][v];
    ldsS[w * 32 + lane] = psum;
    __syncthreads();

    // wave w finalizes output column tile nt = w
    v8f d = zero8;
#pragma unroll
    for (int wp = 0; wp < 4; ++wp) {
        const float* src = &ldsC[(((wp * 4 + w) * 32) + lane) * 8];
#pragma unroll
        for (int v = 0; v < 8; ++v) d[v] += src[v];
    }

#pragma unroll
    for (int v = 0; v < 8; ++v) {
        const int rowM = half * 8 + v;     // C layout: VGPR v holds M = v + 8*half
        float S = 0.f;
#pragma unroll
        for (int wp = 0; wp < 4; ++wp)
            S += ldsS[wp * 32 + rowM] + ldsS[wp * 32 + rowM + 16];
        float val = d[v] / S;              // softmax-weighted aggregation
        val = val > 0.f ? val : (__expf(val) - 1.0f);   // ELU
        out[((size_t)b * N + i0 + rowM) * OUT_F + w * 16 + r] = val;
    }
}

// ---------------------------------------------------------------------------
extern "C" void kernel_launch(void* const* d_in, const int* in_sizes, int n_in,
                              void* d_out, int out_size, void* d_ws, size_t ws_size,
                              hipStream_t stream)
{
    const float* h    = (const float*)d_in[0];
    const int*   adj  = (const int*)  d_in[1];
    const float* W    = (const float*)d_in[2];
    const float* a    = (const float*)d_in[3];
    const float* bias = (const float*)d_in[4];
    float*       out  = (float*)d_out;

    const int N = NN;
    const int B = in_sizes[0] / (N * IN_F);

    char* ws = (char*)d_ws;
    _Float16* whT = (_Float16*)ws;
    size_t off = (size_t)B * OUT_F * N * sizeof(_Float16);
    float* wh1 = (float*)(ws + off); off += (size_t)B * N * sizeof(float);
    float* wh2 = (float*)(ws + off); off += (size_t)B * N * sizeof(float);
    float* m2  = (float*)(ws + off);

    gat_wh_kernel<<<B * N, 64, 0, stream>>>(h, W, a, bias, whT, wh1, wh2, N);
    gat_max_kernel<<<B, 256, 0, stream>>>(wh2, m2, N);
    gat_attn_kernel<<<B * (N / 16), 128, 0, stream>>>(adj, whT, wh1, wh2, m2, out, N);
}
